// AssociativeMemory_24970939859673
// MI455X (gfx1250) — matrix operand order repaired and checked by reading, not compiled, not run on hardware
//
#include <hip/hip_runtime.h>
#include <stdint.h>

typedef __attribute__((ext_vector_type(16))) _Float16 v16h;
typedef __attribute__((ext_vector_type(8)))  _Float16 v8h;
typedef __attribute__((ext_vector_type(8)))  float    v8f;
typedef __attribute__((ext_vector_type(4)))  float    v4f;

#define HDC_D       256
#define NKEYS       4096
#define KB          64
#define NBLK        (NKEYS / KB)
#define ROWS_PER_WG 64
#define NTHREADS    128          // 4 waves of 32
#define NROWS_TOTAL (2 * 8192)
#define LOG2E       1.44269504088896340f

#define USE_ASYNC_COPY 1

// ---------------- CDNA5 helpers ----------------

__device__ __forceinline__ void cp_b128(void* ldst, const void* gsrc) {
#if USE_ASYNC_COPY
  // GLOBAL_LOAD_ASYNC_TO_LDS_B128: VDST = LDS byte address, VADDR = 64-bit global addr.
  // Generic pointers to LDS have the LDS byte offset in their low 32 bits.
  uint32_t l = (uint32_t)(uintptr_t)ldst;
  asm volatile("global_load_async_to_lds_b128 %0, %1, off" :: "v"(l), "v"(gsrc) : "memory");
#else
  *(v8h*)ldst = *(const v8h*)gsrc;
#endif
}

__device__ __forceinline__ void async_wait0() {
#if USE_ASYNC_COPY
  asm volatile("s_wait_asynccnt 0x0" ::: "memory");
#endif
}

// Intra-wave LDS RAW ordering (DS ops of one wave complete in order).
__device__ __forceinline__ void lds_fence() {
  asm volatile("s_wait_dscnt 0x0" ::: "memory");
}

__device__ __forceinline__ void store_fence() {
  asm volatile("s_wait_storecnt 0x0" ::: "memory");
}

__device__ __forceinline__ v8f zero8() {
  v8f z = {0.f, 0.f, 0.f, 0.f, 0.f, 0.f, 0.f, 0.f};
  return z;
}

// A-fragment (16x32 f16, M x K): lane holds one row (M = lane&15); element e maps to
// K = (e&8)*2 + (lane>=16)*8 + (e&7)  -> two contiguous 8-half LDS reads per fragment.
__device__ __forceinline__ v16h load_a_frag(const _Float16* rowp, int k0, int half) {
  v8h lo = *(const v8h*)(rowp + k0 + half * 8);
  v8h hi = *(const v8h*)(rowp + k0 + 16 + half * 8);
  v16h a;
#pragma unroll
  for (int i = 0; i < 8; ++i) { a[i] = lo[i]; a[i + 8] = hi[i]; }
  return a;
}

// B-fragment (32x16 f16, K x N): lane holds column N = lane&15,
// K = (e&8)*2 + (lane>=16)*8 + (e&7) (caller supplies row + k0 + (lane>=16)*16).
__device__ __forceinline__ v16h load_b_frag(const _Float16* p) {
  const int h = (int)(__lane_id() >> 4);
  v8h lo = *(const v8h*)(p - h * 8);
  v8h hi = *(const v8h*)(p + 16 - h * 8);
  v16h b;
#pragma unroll
  for (int i = 0; i < 8; ++i) { b[i] = lo[i]; b[i + 8] = hi[i]; }
  return b;
}

__device__ __forceinline__ v8f wmma16(v16h a, v16h b, v8f c) {
  return __builtin_amdgcn_wmma_f32_16x16x32_f16(false, a, false, b, (short)0, c,
                                                false, false);
}

// ---------------- prep kernels ----------------

__global__ void prep_kv(const float* __restrict__ keys, const float* __restrict__ vals,
                        _Float16* __restrict__ Krow, _Float16* __restrict__ Kt,
                        _Float16* __restrict__ Vt) {
  int idx = blockIdx.x * 256 + threadIdx.x;      // < NKEYS * HDC_D
  int n = idx >> 8, d = idx & 255;
  float k = keys[idx];
  Krow[idx] = (_Float16)k;
  Kt[(size_t)d * NKEYS + n] = (_Float16)k;
  Vt[(size_t)d * NKEYS + n] = (_Float16)vals[idx];
}

__global__ void prep_q(const float* __restrict__ q, _Float16* __restrict__ qh) {
  int idx = blockIdx.x * 256 + threadIdx.x;      // < NROWS_TOTAL * HDC_D
  qh[idx] = (_Float16)q[idx];
}

// ---------------- one Hopfield step: q_out = softmax(beta * q K^T) K ----------------

__global__ __launch_bounds__(NTHREADS) void hopfield_step(
    const _Float16* __restrict__ qin, _Float16* __restrict__ qout,
    const _Float16* __restrict__ Krow, const _Float16* __restrict__ Kt,
    const float* __restrict__ logb) {
  __shared__ alignas(16) _Float16 q_lds[ROWS_PER_WG][264];   // 33 KB (pad 8)
  __shared__ alignas(16) _Float16 k_lds[KB][264];            // 33 KB
  __shared__ alignas(16) _Float16 kt_lds[HDC_D][72];         // 36 KB
  __shared__ alignas(16) float    sc_lds[4][16][68];         // 17 KB
  __shared__ alignas(16) _Float16 p_lds[4][16][72];          // 9 KB
  __shared__ alignas(16) float    row_stat[4][16];

  const int tid  = threadIdx.x;
  const int wave = tid >> 5;
  const int lane = tid & 31;
  const int ll   = lane & 15;
  const int half = lane >> 4;
  const int row0 = blockIdx.x * ROWS_PER_WG;

  const float beta = __builtin_amdgcn_exp2f(logb[0] * LOG2E);

  // stage 64x256 f16 q tile
  for (int c = tid; c < 2048; c += NTHREADS) {
    const int r = c >> 5, o = (c & 31) * 8;
    cp_b128(&q_lds[r][o], qin + (size_t)(row0 + r) * HDC_D + o);
  }
  async_wait0();
  __syncthreads();

  v8f acc[16];
#pragma unroll
  for (int t = 0; t < 16; ++t) acc[t] = zero8();
  float m_r = -1e30f, l_r = 0.0f;

  const _Float16* qrow = &q_lds[wave * 16 + ll][0];
  const _Float16* prow = &p_lds[wave][ll][0];

  for (int kb = 0; kb < NBLK; ++kb) {
    const int n0 = kb * KB;
    __syncthreads();   // previous block's LDS consumers done
    for (int c = tid; c < 2048; c += NTHREADS) {             // 64x256 key rows
      const int r = c >> 5, o = (c & 31) * 8;
      cp_b128(&k_lds[r][o], Krow + (size_t)(n0 + r) * HDC_D + o);
    }
    for (int c = tid; c < 2048; c += NTHREADS) {             // 256x64 key columns
      const int r = c >> 3, o = (c & 7) * 8;
      cp_b128(&kt_lds[r][o], Kt + (size_t)r * NKEYS + n0 + o);
    }
    async_wait0();
    __syncthreads();

    // ---- scores: 16x64 per wave = 4 N-tiles x 8 K-chunks of WMMA ----
#pragma unroll
    for (int nt = 0; nt < 4; ++nt) {
      v8f c = zero8();
      const _Float16* brow = &k_lds[nt * 16 + ll][half * 16];
#pragma unroll
      for (int kt = 0; kt < 8; ++kt) {
        v16h a = load_a_frag(qrow, kt * 32, half);
        v16h b = load_b_frag(brow + kt * 32);
        c = wmma16(a, b, c);
      }
#pragma unroll
      for (int v = 0; v < 8; ++v)
        sc_lds[wave][v + 8 * half][nt * 16 + ll] = c[v] * beta;
    }
    lds_fence();

    // ---- per-wave online softmax: lane owns row ll, columns half*32..+31 ----
    float bm = -1e30f;
#pragma unroll
    for (int j = 0; j < 32; ++j) bm = fmaxf(bm, sc_lds[wave][ll][half * 32 + j]);
    bm = fmaxf(bm, __shfl_xor(bm, 16, 32));
    const float m_new = fmaxf(m_r, bm);
    const float scale = __builtin_amdgcn_exp2f((m_r - m_new) * LOG2E);
    float psum = 0.0f;
#pragma unroll
    for (int j = 0; j < 32; ++j) {
      const float p =
          __builtin_amdgcn_exp2f((sc_lds[wave][ll][half * 32 + j] - m_new) * LOG2E);
      p_lds[wave][ll][half * 32 + j] = (_Float16)p;
      psum += p;
    }
    psum += __shfl_xor(psum, 16, 32);
    l_r = l_r * scale + psum;
    m_r = m_new;
    if (half == 0) row_stat[wave][ll] = scale;
    lds_fence();

    float f[8];
#pragma unroll
    for (int v = 0; v < 8; ++v) f[v] = row_stat[wave][v + 8 * half];

    // ---- q update: acc = acc*scale + p @ K_block (16 D-tiles x 2 K-chunks) ----
#pragma unroll
    for (int dt = 0; dt < 16; ++dt) {
      v8f c = acc[dt];
#pragma unroll
      for (int v = 0; v < 8; ++v) c[v] *= f[v];
      const _Float16* btrow = &kt_lds[dt * 16 + ll][half * 16];
#pragma unroll
      for (int nk = 0; nk < 2; ++nk) {
        v16h a = load_a_frag(prow, nk * 32, half);
        v16h b = load_b_frag(btrow + nk * 32);
        c = wmma16(a, b, c);
      }
      acc[dt] = c;
    }
  }

  // normalize and store q_out (f16)
  if (half == 0) row_stat[wave][ll] = 1.0f / l_r;
  lds_fence();
  float linv[8];
#pragma unroll
  for (int v = 0; v < 8; ++v) linv[v] = row_stat[wave][v + 8 * half];
#pragma unroll
  for (int dt = 0; dt < 16; ++dt) {
#pragma unroll
    for (int v = 0; v < 8; ++v) {
      const int grow = row0 + wave * 16 + v + 8 * half;
      qout[(size_t)grow * HDC_D + dt * 16 + ll] = (_Float16)(acc[dt][v] * linv[v]);
    }
  }
}

// ---------------- final step: attn = softmax(beta q K^T); retrieved = attn V ----------------

__global__ __launch_bounds__(NTHREADS) void final_step(
    const _Float16* __restrict__ qin, const _Float16* __restrict__ Krow,
    const _Float16* __restrict__ Vt, const float* __restrict__ logb,
    float* __restrict__ attn, float* __restrict__ retr) {
  __shared__ alignas(16) _Float16 q_lds[ROWS_PER_WG][264];
  __shared__ alignas(16) _Float16 k_lds[KB][264];
  __shared__ alignas(16) _Float16 vt_lds[HDC_D][72];
  __shared__ alignas(16) float    sc_lds[4][16][68];
  __shared__ alignas(16) _Float16 p_lds[4][16][72];

  const int tid  = threadIdx.x;
  const int wave = tid >> 5;
  const int lane = tid & 31;
  const int ll   = lane & 15;
  const int half = lane >> 4;
  const int row0 = blockIdx.x * ROWS_PER_WG;

  const float beta = __builtin_amdgcn_exp2f(logb[0] * LOG2E);

  for (int c = tid; c < 2048; c += NTHREADS) {
    const int r = c >> 5, o = (c & 31) * 8;
    cp_b128(&q_lds[r][o], qin + (size_t)(row0 + r) * HDC_D + o);
  }
  async_wait0();
  __syncthreads();

  const _Float16* qrow = &q_lds[wave * 16 + ll][0];
  const _Float16* prow = &p_lds[wave][ll][0];
  float m_r = -1e30f, l_r = 0.0f;

  // -------- pass 1: raw scores -> attn buffer, track (m, l) online --------
  for (int kb = 0; kb < NBLK; ++kb) {
    const int n0 = kb * KB;
    __syncthreads();
    for (int c = tid; c < 2048; c += NTHREADS) {
      const int r = c >> 5, o = (c & 31) * 8;
      cp_b128(&k_lds[r][o], Krow + (size_t)(n0 + r) * HDC_D + o);
    }
    async_wait0();
    __syncthreads();

#pragma unroll
    for (int nt = 0; nt < 4; ++nt) {
      v8f c = zero8();
      const _Float16* brow = &k_lds[nt * 16 + ll][half * 16];
#pragma unroll
      for (int kt = 0; kt < 8; ++kt) {
        v16h a = load_a_frag(qrow, kt * 32, half);
        v16h b = load_b_frag(brow + kt * 32);
        c = wmma16(a, b, c);
      }
#pragma unroll
      for (int v = 0; v < 8; ++v) {
        const float s = c[v] * beta;
        sc_lds[wave][v + 8 * half][nt * 16 + ll] = s;
        attn[(size_t)(row0 + wave * 16 + v + 8 * half) * NKEYS + n0 + nt * 16 + ll] = s;
      }
    }
    lds_fence();

    float bm = -1e30f;
#pragma unroll
    for (int j = 0; j < 32; ++j) bm = fmaxf(bm, sc_lds[wave][ll][half * 32 + j]);
    bm = fmaxf(bm, __shfl_xor(bm, 16, 32));
    const float m_new = fmaxf(m_r, bm);
    const float scale = __builtin_amdgcn_exp2f((m_r - m_new) * LOG2E);
    float psum = 0.0f;
#pragma unroll
    for (int j = 0; j < 32; ++j)
      psum += __builtin_amdgcn_exp2f((sc_lds[wave][ll][half * 32 + j] - m_new) * LOG2E);
    psum += __shfl_xor(psum, 16, 32);
    l_r = l_r * scale + psum;
    m_r = m_new;
  }

  store_fence();      // raw score stores globally visible before re-read
  __syncthreads();

  // -------- pass 2: normalize attn in place, accumulate retrieved = attn @ V --------
  const float linv = 1.0f / l_r;
  v8f acc[16];
#pragma unroll
  for (int t = 0; t < 16; ++t) acc[t] = zero8();

  for (int kb = 0; kb < NBLK; ++kb) {
    const int n0 = kb * KB;
    __syncthreads();
    for (int c = tid; c < 2048; c += NTHREADS) {             // 256x64 value columns
      const int r = c >> 3, o = (c & 7) * 8;
      cp_b128(&vt_lds[r][o], Vt + (size_t)r * NKEYS + n0 + o);
    }
    async_wait0();
    __syncthreads();

    float* srow = attn + (size_t)(row0 + wave * 16 + ll) * NKEYS + n0 + half * 32;
#pragma unroll
    for (int j = 0; j < 32; j += 4) {
      v4f s4 = *(const v4f*)(srow + j);
      v4f pn;
#pragma unroll
      for (int q = 0; q < 4; ++q)
        pn[q] = __builtin_amdgcn_exp2f((s4[q] - m_r) * LOG2E) * linv;
      *(v4f*)(srow + j) = pn;                                 // normalized attn out
#pragma unroll
      for (int q = 0; q < 4; ++q)
        p_lds[wave][ll][half * 32 + j + q] = (_Float16)pn[q];
    }
    lds_fence();

#pragma unroll
    for (int dt = 0; dt < 16; ++dt) {
      v8f c = acc[dt];
      const _Float16* btrow = &vt_lds[dt * 16 + ll][half * 16];
#pragma unroll
      for (int nk = 0; nk < 2; ++nk) {
        v16h a = load_a_frag(prow, nk * 32, half);
        v16h b = load_b_frag(btrow + nk * 32);
        c = wmma16(a, b, c);
      }
      acc[dt] = c;
    }
  }

#pragma unroll
  for (int dt = 0; dt < 16; ++dt) {
#pragma unroll
    for (int v = 0; v < 8; ++v) {
      const int grow = row0 + wave * 16 + v + 8 * half;
      retr[(size_t)grow * HDC_D + dt * 16 + ll] = acc[dt][v];
    }
  }
}

// ---------------- launch ----------------

extern "C" void kernel_launch(void* const* d_in, const int* in_sizes, int n_in,
                              void* d_out, int out_size, void* d_ws, size_t ws_size,
                              hipStream_t stream) {
  const float* query = (const float*)d_in[0];   // [2,8192,256]
  const float* keys  = (const float*)d_in[1];   // [4096,256]
  const float* vals  = (const float*)d_in[2];   // [4096,256]
  const float* logb  = (const float*)d_in[3];   // scalar

  float* out  = (float*)d_out;
  float* retr = out;                                        // [2,8192,256]
  float* attn = out + (size_t)NROWS_TOTAL * HDC_D;          // [2,8192,4096]

  // workspace layout (f16): Krow 2MB | Kt 2MB | Vt 2MB | q0 8MB | q1 8MB = 22MB
  _Float16* Krow = (_Float16*)d_ws;
  _Float16* Kt   = Krow + (size_t)NKEYS * HDC_D;
  _Float16* Vt   = Kt   + (size_t)HDC_D * NKEYS;
  _Float16* q0   = Vt   + (size_t)HDC_D * NKEYS;
  _Float16* q1   = q0   + (size_t)NROWS_TOTAL * HDC_D;

  prep_kv<<<(NKEYS * HDC_D) / 256, 256, 0, stream>>>(keys, vals, Krow, Kt, Vt);
  prep_q<<<(NROWS_TOTAL * HDC_D) / 256, 256, 0, stream>>>(query, q0);

  hopfield_step<<<NROWS_TOTAL / ROWS_PER_WG, NTHREADS, 0, stream>>>(q0, q1, Krow, Kt, logb);
  hopfield_step<<<NROWS_TOTAL / ROWS_PER_WG, NTHREADS, 0, stream>>>(q1, q0, Krow, Kt, logb);
  final_step<<<NROWS_TOTAL / ROWS_PER_WG, NTHREADS, 0, stream>>>(q0, Krow, Vt, logb, attn, retr);
}
